// TransformerEmbedding_referenceSet_29420525977874
// MI455X (gfx1250) — compile-verified, matrix operand order checked by reading, not run on hardware
//
#include <hip/hip_runtime.h>
#include <hip/hip_bf16.h>
#include <cmath>

#define BB      64
#define SEQ     512
#define DMODEL  1088
#define NH      8
#define HD      136
#define HDP     160   // q/k head-dim padded to multiple of 32 (WMMA K-step)
#define HDN     144   // v^T N padded to multiple of 16
#define DFF_    2048
#define NLAYER  2

typedef __attribute__((ext_vector_type(16))) __bf16 v16bf;
typedef __attribute__((ext_vector_type(8)))  float  v8f;
typedef __attribute__((ext_vector_type(2)))  float  v2f;
typedef __attribute__((ext_vector_type(2)))  __bf16 v2bf;

// Convert 16 staged fp32 (4x float4) to 16 bf16 and store 32B into LDS.
// Each pair converts via v2f -> v2bf16 (one v_cvt_pk_bf16_f32), already in
// final word layout -> no repack, two ds_store_b128.
__device__ __forceinline__ void stage_tile(unsigned short* dst, const float4* f) {
    union { v2bf b[8]; uint4 q[2]; } pk;
#pragma unroll
    for (int v = 0; v < 4; ++v) {
        v2f lo; lo.x = f[v].x; lo.y = f[v].y;
        v2f hi; hi.x = f[v].z; hi.y = f[v].w;
        pk.b[v*2+0] = __builtin_convertvector(lo, v2bf);
        pk.b[v*2+1] = __builtin_convertvector(hi, v2bf);
    }
    uint4* d = (uint4*)dst;
    d[0] = pk.q[0]; d[1] = pk.q[1];
}

// Load one 16x32 bf16 WMMA operand fragment from an LDS tile (row stride 40
// halves = 80B, 16B aligned). Per ISA 16-bit operand layout: lane L -> row
// (L&15), kbase = (L>>4)*8; elems 0..7 = K kbase..kbase+7, 8..15 = +16.
// Two contiguous 16B chunks -> 2x ds_load_b128.
__device__ __forceinline__ v16bf ld_frag(const unsigned short* p, int lane) {
    int row = lane & 15;
    int kb  = (lane >> 4) << 3;
    const unsigned short* r = p + row * 40 + kb;
    union { uint4 q[2]; v16bf v; } u;
    u.q[0] = *(const uint4*)(r);
    u.q[1] = *(const uint4*)(r + 16);
    return u.v;
}

// Generic batched GEMM: C[z] = alpha * (A[z] (MxK) * W[z]^T (NxK)) + bias [, relu]
// fp32 in / fp32 out, bf16 WMMA compute with fp32 accumulate.
// Block tile 128x128, 8 wave32 waves, each wave 32x64 (2x4 WMMA accums).
// Double-buffered LDS, branchless staging (row indices clamped; stores guarded).
// Requires K % 32 == 0, lda/ldw % 4 == 0 (true for all uses here).
__global__ __launch_bounds__(256)
void gemm_bf16_wmma(const float* __restrict__ A, long long lda, long long sA,
                    const float* __restrict__ W, long long ldw, long long sW,
                    const float* __restrict__ bias,
                    float* __restrict__ C, long long ldc, long long sC,
                    int M, int N, int K, float alpha, int relu)
{
    __shared__ __align__(16) unsigned short As[2][128 * 40];
    __shared__ __align__(16) unsigned short Bs[2][128 * 40];

    int z = blockIdx.z;
    A += (long long)z * sA;
    W += (long long)z * sW;
    C += (long long)z * sC;

    int tid  = threadIdx.x;
    int lane = tid & 31;
    int wave = tid >> 5;
    int wm   = wave >> 1;   // 0..3 -> 32-row slab
    int wn   = wave & 1;    // 0..1 -> 64-col slab
    int m0   = blockIdx.y * 128;
    int n0   = blockIdx.x * 128;

    // Both stagers: 128 rows x 32 K, one thread loads 16 contiguous floats.
    int srow = tid >> 1;            // 0..127
    int scol = (tid & 1) << 4;      // 0 or 16

    // Clamp staging rows: out-of-range rows stage stale data that only feeds
    // never-stored output rows/cols. Removes all exec-mask churn in the loop.
    int gmA = m0 + srow; int gmc = gmA < M ? gmA : (M - 1);
    int gnW = n0 + srow; int gnc = gnW < N ? gnW : (N - 1);
    const float* aRow = A + (long long)gmc * lda + scol;
    const float* wRow = W + (long long)gnc * ldw + scol;

    unsigned short* aDst0 = &As[0][srow * 40 + scol];
    unsigned short* bDst0 = &Bs[0][srow * 40 + scol];
    unsigned short* aDst1 = &As[1][srow * 40 + scol];
    unsigned short* bDst1 = &Bs[1][srow * 40 + scol];

    v8f acc[2][4] = {};
    float4 fa[4], fw[4];

    // Prologue: stage K-slab 0 into buffer 0.
#pragma unroll
    for (int v = 0; v < 4; ++v) {
        fa[v] = ((const float4*)aRow)[v];
        fw[v] = ((const float4*)wRow)[v];
    }
    stage_tile(aDst0, fa);
    stage_tile(bDst0, fw);
    __syncthreads();

    int buf = 0;
    for (int k0 = 0; k0 < K; k0 += 32) {
        bool more = (k0 + 32) < K;            // wave-uniform
        if (more) {
            // Issue next slab's global loads before the WMMAs.
#pragma unroll
            for (int v = 0; v < 4; ++v) {
                fa[v] = ((const float4*)(aRow + k0 + 32))[v];
                fw[v] = ((const float4*)(wRow + k0 + 32))[v];
            }
            if (k0 + 64 < K) {
                __builtin_prefetch(aRow + k0 + 64, 0, 1);
                __builtin_prefetch(wRow + k0 + 64, 0, 1);
            }
        }

        const unsigned short* aCur = As[buf];
        const unsigned short* bCur = Bs[buf];
        v16bf af[2], bf[4];
#pragma unroll
        for (int i = 0; i < 2; ++i)
            af[i] = ld_frag(&aCur[(wm * 32 + i * 16) * 40], lane);
#pragma unroll
        for (int j = 0; j < 4; ++j)
            bf[j] = ld_frag(&bCur[(wn * 64 + j * 16) * 40], lane);

#pragma unroll
        for (int i = 0; i < 2; ++i)
#pragma unroll
            for (int j = 0; j < 4; ++j)
                acc[i][j] = __builtin_amdgcn_wmma_f32_16x16x32_bf16(
                    false, af[i], false, bf[j], (short)0, acc[i][j], false, false);

        if (more) {
            stage_tile(buf ? aDst0 : aDst1, fa);
            stage_tile(buf ? bDst0 : bDst1, fw);
            __syncthreads();
            buf ^= 1;
        }
    }

    // C fragment layout: VGPR r, lanes 0-15 -> M=r, lanes 16-31 -> M=r+8, N=lane&15
    int mlane = (lane >> 4) << 3;
    int nlane = lane & 15;
#pragma unroll
    for (int i = 0; i < 2; ++i) {
#pragma unroll
        for (int j = 0; j < 4; ++j) {
            int gn = n0 + wn * 64 + j * 16 + nlane;
            if (gn < N) {
                float bb = bias ? bias[gn] : 0.0f;
#pragma unroll
                for (int r = 0; r < 8; ++r) {
                    int gm = m0 + wm * 32 + i * 16 + r + mlane;
                    if (gm < M) {
                        float v = acc[i][j][r] * alpha + bb;
                        if (relu) v = v > 0.0f ? v : 0.0f;
                        C[(long long)gm * ldc + gn] = v;
                    }
                }
            }
        }
    }
}

// Assemble S (B,512,1088) and mask bias (B,512)
__global__ void build_s_kernel(const float* __restrict__ q, const float* __restrict__ sa,
                               const float* __restrict__ si, const int* __restrict__ na,
                               const int* __restrict__ ni, const float* __restrict__ ref,
                               float* __restrict__ S, float* __restrict__ maskbias)
{
    int tok = blockIdx.x;          // b*512 + j
    int b = tok >> 9;
    int j = tok & 511;
    float* out = S + (long long)tok * DMODEL;
    const float* src; float tail;
    if (j == 0)      { src = q   + (long long)b * 1024;                  tail =  0.0f; }
    else if (j < 13) { src = sa  + ((long long)b * 12 + (j - 1)) * 1024; tail =  1.0f; }
    else if (j < 25) { src = si  + ((long long)b * 12 + (j - 13)) * 1024; tail = -1.0f; }
    else             { src = ref + ((long long)b * 487 + (j - 25)) * 1024; tail = 0.0f; }
    for (int d = threadIdx.x; d < DMODEL; d += blockDim.x)
        out[d] = (d < 1024) ? src[d] : tail;
    if (threadIdx.x == 0) {
        int ra = na[b], ri = ni[b];
        bool m = (j == 0) || (j >= 1 && j < 13 && j < ra + 1) ||
                 (j >= 13 && j < 25 && j < ri + 13) || (j >= 25 && j < ri + 13);
        maskbias[tok] = m ? -1.0e9f : 0.0f;
    }
}

// qkv (B,512,3264) -> qp/kp (B*H,512,160) zero-padded, vt (B*H,144,512)
__global__ void repack_qkv_kernel(const float* __restrict__ qkv, float* __restrict__ qp,
                                  float* __restrict__ kp, float* __restrict__ vt)
{
    long long idx = blockIdx.x;    // z*512 + s, z = b*NH + h
    int s  = (int)(idx & 511);
    int z  = (int)(idx >> 9);
    int h  = z & 7;
    int bt = z >> 3;
    const float* base = qkv + ((long long)(bt * SEQ + s)) * (3 * DMODEL) + h * HD;
    float* qo = qp + ((long long)z * SEQ + s) * HDP;
    float* ko = kp + ((long long)z * SEQ + s) * HDP;
    float* vo = vt + (long long)z * HDN * SEQ + s;
    for (int d = threadIdx.x; d < HDP; d += blockDim.x) {
        float qv = (d < HD) ? base[d] : 0.0f;
        float kv = (d < HD) ? base[DMODEL + d] : 0.0f;
        qo[d] = qv; ko[d] = kv;
        if (d < HDN) vo[(long long)d * SEQ] = (d < HD) ? base[2 * DMODEL + d] : 0.0f;
    }
}

__global__ __launch_bounds__(256)
void softmax_kernel(float* __restrict__ scores, const float* __restrict__ maskbias)
{
    __shared__ float red[256];
    long long row = blockIdx.x;            // (b*NH+h)*512 + q
    int z = (int)(row >> 9);
    int b = z >> 3;
    float* r = scores + row * SEQ;
    const float* mb = maskbias + (long long)b * SEQ;
    int t = threadIdx.x;
    float e0 = r[t]       + mb[t];
    float e1 = r[t + 256] + mb[t + 256];
    red[t] = fmaxf(e0, e1); __syncthreads();
    for (int o = 128; o > 0; o >>= 1) { if (t < o) red[t] = fmaxf(red[t], red[t + o]); __syncthreads(); }
    float mx = red[0]; __syncthreads();
    e0 = __expf(e0 - mx); e1 = __expf(e1 - mx);
    red[t] = e0 + e1; __syncthreads();
    for (int o = 128; o > 0; o >>= 1) { if (t < o) red[t] += red[t + o]; __syncthreads(); }
    float inv = 1.0f / red[0];
    r[t] = e0 * inv; r[t + 256] = e1 * inv;
}

// ov (B*H,512,144) -> ctx (B,512,1088)
__global__ void gather_ctx_kernel(const float* __restrict__ ov, float* __restrict__ ctx)
{
    long long tok = blockIdx.x;   // b*512 + s
    int s = (int)(tok & 511);
    int b = (int)(tok >> 9);
    float* o = ctx + tok * DMODEL;
    for (int i = threadIdx.x; i < DMODEL; i += blockDim.x) {
        int h = i / HD, d = i - h * HD;
        o[i] = ov[(((long long)(b * NH + h) * SEQ) + s) * HDN + d];
    }
}

// x = LayerNorm(x + o) * w + b   (in place on x)
__global__ __launch_bounds__(256)
void add_ln_kernel(float* __restrict__ x, const float* __restrict__ o,
                   const float* __restrict__ w, const float* __restrict__ bq)
{
    __shared__ float red[256];
    long long tok = blockIdx.x;
    float* xr = x + tok * DMODEL;
    const float* orr = o + tok * DMODEL;
    int t = threadIdx.x;
    float v[5];
    float s = 0.0f;
#pragma unroll
    for (int i = 0; i < 5; ++i) {
        int d = t + i * 256;
        v[i] = (d < DMODEL) ? (xr[d] + orr[d]) : 0.0f;
        s += v[i];
    }
    red[t] = s; __syncthreads();
    for (int off = 128; off > 0; off >>= 1) { if (t < off) red[t] += red[t + off]; __syncthreads(); }
    float mu = red[0] / (float)DMODEL; __syncthreads();
    s = 0.0f;
#pragma unroll
    for (int i = 0; i < 5; ++i) {
        int d = t + i * 256;
        float dv = (d < DMODEL) ? (v[i] - mu) : 0.0f;
        s += dv * dv;
    }
    red[t] = s; __syncthreads();
    for (int off = 128; off > 0; off >>= 1) { if (t < off) red[t] += red[t + off]; __syncthreads(); }
    float rs = rsqrtf(red[0] / (float)DMODEL + 1e-5f);
#pragma unroll
    for (int i = 0; i < 5; ++i) {
        int d = t + i * 256;
        if (d < DMODEL) xr[d] = (v[i] - mu) * rs * w[d] + bq[d];
    }
}

// out = concat(q_out 64x1x1024, sa_out 64x12x1024, si_out 64x12x1024)
__global__ void extract_out_kernel(const float* __restrict__ x, float* __restrict__ out)
{
    long long i = (long long)blockIdx.x * blockDim.x + threadIdx.x;
    const long long QN = 64LL * 1024, SAN = 64LL * 12 * 1024;
    if (i >= QN + 2 * SAN) return;
    int b, tokj, d;
    if (i < QN) { b = (int)(i >> 10); tokj = 0; d = (int)(i & 1023); }
    else if (i < QN + SAN) {
        long long r = i - QN; b = (int)(r / (12 * 1024));
        long long rr = r % (12 * 1024); tokj = 1 + (int)(rr >> 10); d = (int)(rr & 1023);
    } else {
        long long r = i - QN - SAN; b = (int)(r / (12 * 1024));
        long long rr = r % (12 * 1024); tokj = 13 + (int)(rr >> 10); d = (int)(rr & 1023);
    }
    out[i] = x[((long long)b * SEQ + tokj) * DMODEL + d];
}

extern "C" void kernel_launch(void* const* d_in, const int* in_sizes, int n_in,
                              void* d_out, int out_size, void* d_ws, size_t ws_size,
                              hipStream_t stream)
{
    const float* q_emb  = (const float*)d_in[0];
    const float* sa_emb = (const float*)d_in[1];
    const float* si_emb = (const float*)d_in[2];
    const int*   nA     = (const int*)d_in[3];
    const int*   nI     = (const int*)d_in[4];
    const float* ref    = (const float*)d_in[5];
    const float* Wqkv   = (const float*)d_in[6];
    const float* bqkv   = (const float*)d_in[7];
    const float* Wo     = (const float*)d_in[8];
    const float* bo     = (const float*)d_in[9];
    const float* ln1w   = (const float*)d_in[10];
    const float* ln1b   = (const float*)d_in[11];
    const float* W1     = (const float*)d_in[12];
    const float* b1     = (const float*)d_in[13];
    const float* W2     = (const float*)d_in[14];
    const float* b2     = (const float*)d_in[15];
    const float* ln2w   = (const float*)d_in[16];
    const float* ln2b   = (const float*)d_in[17];

    float* ws = (float*)d_ws;
    size_t off = 0;
    float* X   = ws + off; off += (size_t)BB * SEQ * DMODEL;        // activations
    float* QKV = ws + off; off += (size_t)BB * SEQ * 3 * DMODEL;    // qkv / ffn hidden
    float* QP  = ws + off; off += (size_t)BB * NH * SEQ * HDP;
    float* KP  = ws + off; off += (size_t)BB * NH * SEQ * HDP;
    float* VT  = ws + off; off += (size_t)BB * NH * HDN * SEQ;
    float* SC  = ws + off; off += (size_t)BB * NH * SEQ * SEQ;      // scores / probs
    float* OV  = ws + off; off += (size_t)BB * NH * SEQ * HDN;
    float* CTX = ws + off; off += (size_t)BB * SEQ * DMODEL;        // ctx / ffn out
    float* PRJ = ws + off; off += (size_t)BB * SEQ * DMODEL;
    float* MB  = ws + off; off += (size_t)BB * SEQ;
    (void)ws_size; (void)in_sizes; (void)n_in; (void)out_size;

    const int Mtok = BB * SEQ;
    const float rscale = 1.0f / sqrtf((float)HD);

    build_s_kernel<<<BB * SEQ, 256, 0, stream>>>(q_emb, sa_emb, si_emb, nA, nI, ref, X, MB);

    for (int l = 0; l < NLAYER; ++l) {
        // qkv = X @ Wqkv[l]^T + bqkv[l]
        {
            dim3 g((3 * DMODEL + 127) / 128, (Mtok + 127) / 128, 1);
            gemm_bf16_wmma<<<g, 256, 0, stream>>>(
                X, DMODEL, 0, Wqkv + (size_t)l * 3 * DMODEL * DMODEL, DMODEL, 0,
                bqkv + (size_t)l * 3 * DMODEL, QKV, 3 * DMODEL, 0,
                Mtok, 3 * DMODEL, DMODEL, 1.0f, 0);
        }
        repack_qkv_kernel<<<BB * NH * SEQ, 256, 0, stream>>>(QKV, QP, KP, VT);
        // scores = (q @ k^T) / sqrt(hd), batched over B*H
        {
            dim3 g(SEQ / 128, SEQ / 128, BB * NH);
            gemm_bf16_wmma<<<g, 256, 0, stream>>>(
                QP, HDP, (long long)SEQ * HDP, KP, HDP, (long long)SEQ * HDP,
                nullptr, SC, SEQ, (long long)SEQ * SEQ,
                SEQ, SEQ, HDP, rscale, 0);
        }
        softmax_kernel<<<BB * NH * SEQ, 256, 0, stream>>>(SC, MB);
        // ov = attn @ v  (W = v^T, NxK = 144x512)
        {
            dim3 g((HDN + 127) / 128, SEQ / 128, BB * NH);
            gemm_bf16_wmma<<<g, 256, 0, stream>>>(
                SC, SEQ, (long long)SEQ * SEQ, VT, SEQ, (long long)HDN * SEQ,
                nullptr, OV, HDN, (long long)SEQ * HDN,
                SEQ, HDN, SEQ, 1.0f, 0);
        }
        gather_ctx_kernel<<<BB * SEQ, 256, 0, stream>>>(OV, CTX);
        // proj = ctx @ Wo[l]^T + bo[l]
        {
            dim3 g((DMODEL + 127) / 128, (Mtok + 127) / 128, 1);
            gemm_bf16_wmma<<<g, 256, 0, stream>>>(
                CTX, DMODEL, 0, Wo + (size_t)l * DMODEL * DMODEL, DMODEL, 0,
                bo + (size_t)l * DMODEL, PRJ, DMODEL, 0,
                Mtok, DMODEL, DMODEL, 1.0f, 0);
        }
        add_ln_kernel<<<Mtok, 256, 0, stream>>>(X, PRJ, ln1w + (size_t)l * DMODEL, ln1b + (size_t)l * DMODEL);
        // h = relu(X @ W1^T + b1)
        {
            dim3 g((DFF_ + 127) / 128, (Mtok + 127) / 128, 1);
            gemm_bf16_wmma<<<g, 256, 0, stream>>>(
                X, DMODEL, 0, W1 + (size_t)l * DFF_ * DMODEL, DMODEL, 0,
                b1 + (size_t)l * DFF_, QKV, DFF_, 0,
                Mtok, DFF_, DMODEL, 1.0f, 1);
        }
        // ffn = h @ W2^T + b2
        {
            dim3 g((DMODEL + 127) / 128, (Mtok + 127) / 128, 1);
            gemm_bf16_wmma<<<g, 256, 0, stream>>>(
                QKV, DFF_, 0, W2 + (size_t)l * DMODEL * DFF_, DFF_, 0,
                b2 + (size_t)l * DMODEL, CTX, DMODEL, 0,
                Mtok, DMODEL, DFF_, 1.0f, 0);
        }
        add_ln_kernel<<<Mtok, 256, 0, stream>>>(X, CTX, ln2w + (size_t)l * DMODEL, ln2b + (size_t)l * DMODEL);
    }

    {
        long long tot = 64LL * 25 * 1024;
        extract_out_kernel<<<(unsigned)((tot + 255) / 256), 256, 0, stream>>>(X, (float*)d_out);
    }
}
